// Conv_78743930404939
// MI455X (gfx1250) — compile-verified, matrix operand order checked by reading, not run on hardware
//
#include <hip/hip_runtime.h>
#include <math.h>

// ---------------------------------------------------------------------------
// FFT-based 3D circular convolution for x:(16,32,128,128), w:(32,128,128).
// Each 1-D FFT stage = matmul by a symmetric DFT matrix, run as complex GEMM
// on v_wmma_f32_16x16x32_f16 with split-f16 error compensation (hi+lo f16
// ~= 22-bit mantissa, f32 accumulation). GEMM operands live as hi/lo f16
// planes; B-consumed buffers use a K-pair-interleaved layout so fragments
// load as 8 dwords with immediate offsets and zero packing VALU.
// ---------------------------------------------------------------------------

typedef _Float16 h16;
typedef __attribute__((ext_vector_type(16))) _Float16 v16h;
typedef __attribute__((ext_vector_type(8)))  _Float16 v8h;
typedef __attribute__((ext_vector_type(8)))  float    v8f;
typedef __attribute__((ext_vector_type(8)))  unsigned v8u;

#define PI_F 3.14159265358979323846f

#define CM_PLAIN 0   // C element addr = bb*cBatch + m*CRS + n
#define CM_MPAIR 1   // C addr = bb*cBatch + (m>>1)*2*CRS + n*2 + (m&1)
#define CM_BPAIR 2   // C addr = (bb>>1)*2*cBatch + (m*CRS+n)*2 + (bb&1)

// A-side fragment: reduction axis contiguous -> two 16-byte vector loads.
__device__ __forceinline__ void loadA(const h16* __restrict__ p, unsigned off, v16h& f) {
    v8h a = *(const v8h*)(p + off);        // e=0..7  : k = kb+grp*8+0..7
    v8h b = *(const v8h*)(p + off + 16);   // e=8..15 : k = kb+16+grp*8+0..7
#pragma unroll
    for (int e = 0; e < 8; ++e) { f[e] = a[e]; f[e + 8] = b[e]; }
}

// B-side fragment from K-pair-interleaved storage: 8 dword loads, immediate
// offsets kp*PRS with kp in {0,1,2,3,8,9,10,11}; (k even -> low half) matches
// the v16h element packing directly (no cross-half moves).
template<int PRS>   // halfs per k-pair row (= 2*row stride)
__device__ __forceinline__ void loadB_pk(const h16* __restrict__ p, unsigned off, v16h& f) {
    v8u u;
#pragma unroll
    for (int ep = 0; ep < 8; ++ep) {
        const unsigned kp = (unsigned)(((ep & 4) << 1) + (ep & 3));
        u[ep] = *(const unsigned*)(p + off + kp * (unsigned)PRS);
    }
    f = __builtin_bit_cast(v16h, u);
}

// Error-compensated real product-accumulate: acc += (ahi+alo)*(bhi+blo)
__device__ __forceinline__ v8f wmma3(v16h ahi, v16h alo, v16h bhi, v16h blo, v8f acc) {
    acc = __builtin_amdgcn_wmma_f32_16x16x32_f16(false, ahi, false, bhi, (short)0, acc, false, false);
    acc = __builtin_amdgcn_wmma_f32_16x16x32_f16(false, ahi, false, blo, (short)0, acc, false, false);
    acc = __builtin_amdgcn_wmma_f32_16x16x32_f16(false, alo, false, bhi, (short)0, acc, false, false);
    return acc;
}

__device__ __forceinline__ void split_store(h16* __restrict__ hi, h16* __restrict__ lo,
                                            unsigned idx, float v) {
    h16 h = (h16)v;
    hi[idx] = h;
    lo[idx] = (h16)(v - (float)h);
}

// Split-store two adjacent values as one packed dword per plane (idx even).
__device__ __forceinline__ void split_store_pk(h16* __restrict__ hi, h16* __restrict__ lo,
                                               unsigned idx, float a, float b) {
    h16 ah = (h16)a, bh = (h16)b;
    h16 al = (h16)(a - (float)ah), bl = (h16)(b - (float)bh);
    union { h16 h[2]; unsigned u; } H, L;
    H.h[0] = ah; H.h[1] = bh; L.h[0] = al; L.h[1] = bl;
    *(unsigned*)(hi + idx) = H.u;
    *(unsigned*)(lo + idx) = L.u;
}

// Strided batched complex GEMM:  C[b] = A[b] * B[b]
//   A: M x K  contiguous in k (element = bb*aBatch + m*ars + k), split planes
//   B: K x N  k-pair-interleaved (pair row stride PRS halfs), always complex
//   C: M x N  layout per CMODE
template<int PRS, int CRS, int CMODE, bool AIMAG, bool WANTIM>
__global__ __launch_bounds__(256) void cgemm_kernel(
    const h16* __restrict__ aReH, const h16* __restrict__ aReL,
    const h16* __restrict__ aImH, const h16* __restrict__ aImL,
    const h16* __restrict__ bReH, const h16* __restrict__ bReL,
    const h16* __restrict__ bImH, const h16* __restrict__ bImL,
    h16* __restrict__ cReH, h16* __restrict__ cReL,
    h16* __restrict__ cImH, h16* __restrict__ cImL,
    float* __restrict__ outRe,
    int K, int tilesM, int tilesN, int nBatch,
    unsigned ars, unsigned aBatch, unsigned bBatch, unsigned cBatch)
{
    int wid   = (int)blockIdx.x * 8 + ((int)threadIdx.x >> 5);
    int total = tilesM * tilesN * nBatch;
    if (wid >= total) return;                    // whole-wave uniform exit

    int tn   = wid % tilesN;
    int rest = wid / tilesN;
    int tm   = rest % tilesM;
    int bb   = rest / tilesM;

    int lane = (int)threadIdx.x & 31;
    int grp  = lane >> 4;
    int lr   = lane & 15;

    unsigned mBase = (unsigned)(tm << 4);
    unsigned nBase = (unsigned)(tn << 4);

    unsigned aK = (unsigned)bb * aBatch + (mBase + (unsigned)lr) * ars + (unsigned)(grp << 3);
    unsigned bK = (unsigned)bb * bBatch + (nBase + (unsigned)lr) * 2u
                + (unsigned)(grp << 2) * (unsigned)PRS;

    v8f cr = {};     // + Are*Bre
    v8f cs = {};     // + Aim*Bim   (subtracted once at the end)
    v8f ci = {};     // + Are*Bim + Aim*Bre

    for (int kb = 0; kb < K; kb += 32) {
        v16h areH, areL, breH, breL, bimH, bimL;
        loadA(aReH, aK, areH);
        loadA(aReL, aK, areL);
        loadB_pk<PRS>(bReH, bK, breH);
        loadB_pk<PRS>(bReL, bK, breL);
        loadB_pk<PRS>(bImH, bK, bimH);
        loadB_pk<PRS>(bImL, bK, bimL);

        cr = wmma3(areH, areL, breH, breL, cr);
        if (WANTIM) ci = wmma3(areH, areL, bimH, bimL, ci);

        if (AIMAG) {
            v16h aimH, aimL;
            loadA(aImH, aK, aimH);
            loadA(aImL, aK, aimL);
            cs = wmma3(aimH, aimL, bimH, bimL, cs);
            if (WANTIM) ci = wmma3(aimH, aimL, breH, breL, ci);
        }
        aK += 32u;
        bK += 16u * (unsigned)PRS;               // 32 k = 16 pairs
    }

    // ---- store C (D layout: m = mBase + grp*8 + r, n = nBase + lr) -------
    if (!WANTIM) {
        unsigned cOff = (unsigned)bb * cBatch + (mBase + (unsigned)(grp << 3)) * (unsigned)CRS
                      + (nBase + (unsigned)lr);
#pragma unroll
        for (int r = 0; r < 8; ++r)
            outRe[cOff + (unsigned)(r * CRS)] = cr[r] - cs[r];
    } else if (CMODE == CM_PLAIN) {
        unsigned cOff = (unsigned)bb * cBatch + (mBase + (unsigned)(grp << 3)) * (unsigned)CRS
                      + (nBase + (unsigned)lr);
#pragma unroll
        for (int r = 0; r < 8; ++r) {
            unsigned idx = cOff + (unsigned)(r * CRS);
            split_store(cReH, cReL, idx, AIMAG ? (cr[r] - cs[r]) : cr[r]);
            split_store(cImH, cImL, idx, ci[r]);
        }
    } else if (CMODE == CM_MPAIR) {
        unsigned cOff = (unsigned)bb * cBatch + (mBase + (unsigned)(grp << 3)) * (unsigned)CRS
                      + (nBase + (unsigned)lr) * 2u;
#pragma unroll
        for (int rp = 0; rp < 4; ++rp) {
            unsigned idx = cOff + (unsigned)(rp * 2 * CRS);
            float r0 = AIMAG ? (cr[2*rp]   - cs[2*rp])   : cr[2*rp];
            float r1 = AIMAG ? (cr[2*rp+1] - cs[2*rp+1]) : cr[2*rp+1];
            split_store_pk(cReH, cReL, idx, r0, r1);
            split_store_pk(cImH, cImL, idx, ci[2*rp], ci[2*rp+1]);
        }
    } else {   // CM_BPAIR
        unsigned cOff = (unsigned)(bb >> 1) * (2u * cBatch) + (unsigned)(bb & 1)
                      + ((mBase + (unsigned)(grp << 3)) * (unsigned)CRS + nBase + (unsigned)lr) * 2u;
#pragma unroll
        for (int r = 0; r < 8; ++r) {
            unsigned idx = cOff + (unsigned)(r * 2 * CRS);
            split_store(cReH, cReL, idx, AIMAG ? (cr[r] - cs[r]) : cr[r]);
            split_store(cImH, cImL, idx, ci[r]);
        }
    }
}

// Split an f32 array into hi/lo f16 planes (used once for x and w_real).
__global__ void split_f32_kernel(const float* __restrict__ src,
                                 h16* __restrict__ hi, h16* __restrict__ lo, int n)
{
    int i = blockIdx.x * blockDim.x + threadIdx.x;
    if (i >= n) return;
    split_store(hi, lo, (unsigned)i, src[i]);
}

// DFT matrices as split-f16 planes [reH|reL|imH|imL].
// 128-pt tables written twice: plain (A-side) and k-pair interleaved (B-side).
// F_N[r,c] = exp(-2*pi*i*r*c/N); inverse tables carry the 1/N.
__global__ void init_dft_tables(h16* f128p, h16* f128k, h16* i128p, h16* i128k,
                                h16* f32t, h16* i32t)
{
    int t = blockIdx.x * blockDim.x + threadIdx.x;
    if (t < 128 * 128) {
        const int nn = 128 * 128;
        int r = t >> 7, c = t & 127;
        float ang = -2.0f * PI_F * (float)((r * c) & 127) / 128.0f;
        float s, co;
        sincosf(ang, &s, &co);
        float ico = co * (1.0f/128.0f), isn = -s * (1.0f/128.0f);
        int tp = ((r >> 1) << 8) + (c << 1) + (r & 1);     // pk index
        split_store(f128p,        f128p + nn,   t,  co);
        split_store(f128p + 2*nn, f128p + 3*nn, t,  s);
        split_store(f128k,        f128k + nn,   tp, co);
        split_store(f128k + 2*nn, f128k + 3*nn, tp, s);
        split_store(i128p,        i128p + nn,   t,  ico);
        split_store(i128p + 2*nn, i128p + 3*nn, t,  isn);
        split_store(i128k,        i128k + nn,   tp, ico);
        split_store(i128k + 2*nn, i128k + 3*nn, tp, isn);
    }
    if (t < 32 * 32) {
        const int nn = 32 * 32;
        int r = t >> 5, c = t & 31;
        float ang = -2.0f * PI_F * (float)((r * c) & 31) / 32.0f;
        float s, co;
        sincosf(ang, &s, &co);
        split_store(f32t,        f32t + nn,   t, co);
        split_store(f32t + 2*nn, f32t + 3*nn, t, s);
        split_store(i32t,        i32t + nn,   t, co * (1.0f/32.0f));
        split_store(i32t + 2*nn, i32t + 3*nn, t, -s * (1.0f/32.0f));
    }
}

// Pointwise spectral product on split planes (plain layout): X *= W * scale.
__global__ void cmul_split_kernel(h16* __restrict__ xrh, h16* __restrict__ xrl,
                                  h16* __restrict__ xih, h16* __restrict__ xil,
                                  const h16* __restrict__ wrh, const h16* __restrict__ wrl,
                                  const h16* __restrict__ wih, const h16* __restrict__ wil,
                                  float scale, int n, int wmask)
{
    int i = blockIdx.x * blockDim.x + threadIdx.x;
    if (i >= n) return;
    unsigned iw = (unsigned)(i & wmask);
    float wr = ((float)wrh[iw] + (float)wrl[iw]) * scale;
    float wi = ((float)wih[iw] + (float)wil[iw]) * scale;
    float xr = (float)xrh[i] + (float)xrl[i];
    float xi = (float)xih[i] + (float)xil[i];
    split_store(xrh, xrl, (unsigned)i, xr * wr - xi * wi);
    split_store(xih, xil, (unsigned)i, xr * wi + xi * wr);
}

// Complex buffers: 4 consecutive planes [reH reL imH imL] of planeSize halfs.
template<int PRS, int CRS, int CMODE, bool AIMAG, bool WANTIM>
static inline void run_gemm(hipStream_t s,
    const h16* A, long aP,          // A planes (2 if real, 4 if complex)
    const h16* B, long bP,          // B planes (4), k-pair-interleaved
    h16* C, long cP,                // C planes (4), ignored if !WANTIM
    float* outRe,
    int K, int tilesM, int tilesN, int nBatch,
    unsigned ars, unsigned aBatch, unsigned bBatch, unsigned cBatch)
{
    int waves  = tilesM * tilesN * nBatch;
    int blocks = (waves + 7) >> 3;                 // 8 waves (256 thr) / block
    cgemm_kernel<PRS, CRS, CMODE, AIMAG, WANTIM><<<dim3((unsigned)blocks), dim3(256), 0, s>>>(
        A, A + aP,
        AIMAG ? A + 2*aP : nullptr, AIMAG ? A + 3*aP : nullptr,
        B, B + bP, B + 2*bP, B + 3*bP,
        WANTIM ? C : nullptr, WANTIM ? C + cP : nullptr,
        WANTIM ? C + 2*cP : nullptr, WANTIM ? C + 3*cP : nullptr,
        outRe, K, tilesM, tilesN, nBatch, ars, aBatch, bBatch, cBatch);
}

extern "C" void kernel_launch(void* const* d_in, const int* in_sizes, int n_in,
                              void* d_out, int out_size, void* d_ws, size_t ws_size,
                              hipStream_t stream)
{
    const float* x      = (const float*)d_in[0];   // (16,32,128,128)
    const float* w_real = (const float*)d_in[1];   // (32,128,128)
    float*       out    = (float*)d_out;
    h16*         ws     = (h16*)d_ws;

    const long NX = 16L * 32 * 128 * 128;          // 8388608
    const long NW = 32L * 128 * 128;               // 524288

    h16* xs    = ws;                 // 2 planes  (x split, real)
    h16* bufA  = xs    + 2 * NX;     // 4 planes
    h16* bufB  = bufA  + 4 * NX;     // 4 planes
    h16* wsrc  = bufB  + 4 * NX;     // 2 planes
    h16* wA    = wsrc  + 2 * NW;     // 4 planes
    h16* wB    = wA    + 4 * NW;     // 4 planes
    h16* F128p = wB    + 4 * NW;     // 4 * 16384 (plain, A-side)
    h16* F128k = F128p + 4 * 16384;  // 4 * 16384 (k-pair, B-side)
    h16* I128p = F128k + 4 * 16384;
    h16* I128k = I128p + 4 * 16384;
    h16* F32t  = I128k + 4 * 16384;  // 4 * 1024 (plain)
    h16* I32t  = F32t  + 4 * 1024;

    init_dft_tables<<<dim3(64), dim3(256), 0, stream>>>(F128p, F128k, I128p, I128k, F32t, I32t);
    split_f32_kernel<<<dim3(32768), dim3(256), 0, stream>>>(x, xs, xs + NX, (int)NX);
    split_f32_kernel<<<dim3(2048),  dim3(256), 0, stream>>>(w_real, wsrc, wsrc + NW, (int)NW);

    // ---- W = fftn(w_real) ------------------------------------------------
    run_gemm<256, 128, CM_MPAIR, false, true>(stream, wsrc, NW, F128k, 16384, wA, NW, nullptr,
        128, 256, 8, 1,    128, 0, 0, 0);
    run_gemm<256, 128, CM_BPAIR, true, true>(stream, F128p, 16384, wA, NW, wB, NW, nullptr,
        128, 8, 8, 32,     128, 0, 16384, 16384);
    run_gemm<32768, 16384, CM_PLAIN, true, true>(stream, F32t, 1024, wB, NW, wA, NW, nullptr,
        32, 2, 1024, 1,    32, 0, 0, 0);

    // ---- X = fftn(x) -----------------------------------------------------
    run_gemm<256, 128, CM_MPAIR, false, true>(stream, xs, NX, F128k, 16384, bufA, NX, nullptr,
        128, 4096, 8, 1,   128, 0, 0, 0);
    run_gemm<256, 128, CM_BPAIR, true, true>(stream, F128p, 16384, bufA, NX, bufB, NX, nullptr,
        128, 8, 8, 512,    128, 0, 16384, 16384);
    run_gemm<32768, 16384, CM_PLAIN, true, true>(stream, F32t, 1024, bufB, NX, bufA, NX, nullptr,
        32, 2, 1024, 16,   32, 0, 524288, 524288);

    // ---- X *= W / scale  (scale = sqrt(32*128*128) = 512*sqrt(2)) --------
    const float invScale = (float)(1.4142135623730951 / 1024.0);
    cmul_split_kernel<<<dim3(32768), dim3(256), 0, stream>>>(
        bufA, bufA + NX, bufA + 2*NX, bufA + 3*NX,
        wA,   wA   + NW, wA   + 2*NW, wA   + 3*NW,
        invScale, (int)NX, (int)(NW - 1));

    // ---- out = real(ifftn(X)) ; 1/N folded into inverse tables -----------
    run_gemm<256, 128, CM_MPAIR, true, true>(stream, bufA, NX, I128k, 16384, bufB, NX, nullptr,
        128, 4096, 8, 1,   128, 0, 0, 0);
    run_gemm<256, 128, CM_BPAIR, true, true>(stream, I128p, 16384, bufB, NX, bufA, NX, nullptr,
        128, 8, 8, 512,    128, 0, 16384, 16384);
    run_gemm<32768, 16384, CM_PLAIN, true, false>(stream, I32t, 1024, bufA, NX, nullptr, 0, out,
        32, 2, 1024, 16,   32, 0, 524288, 524288);
}
